// MLP_13245679140988
// MI455X (gfx1250) — matrix operand order repaired and checked by reading, not compiled, not run on hardware
//
#include <hip/hip_runtime.h>

// ---------------------------------------------------------------------------
// Fused TriOD-masked MLP for MI455X (gfx1250, wave32, WMMA).
//   y = (ReLU(LN_head(x) @ (Wfc*m_fc)^T)^2) @ (Wproj*m_proj)^T
// Mask structure exploited:
//   m_fc[i,j]   = (4j <= i) && (j < 768) && (i < 3072)
//   m_proj[e,f] = (f <= 4e) && (f < 3072) && (e < 768)
// Pass 1 (prep): masked weights -> bf16 in d_ws (needs 16 MB scratch).
// Pass 2 (fused): LN -> GEMM1(bf16 WMMA) -> ReLU^2 -> GEMM2(bf16 WMMA).
// ---------------------------------------------------------------------------

typedef __attribute__((ext_vector_type(16))) __bf16 v16bf;
typedef __attribute__((ext_vector_type(8)))  float  v8f;

static constexpr int E     = 1024;
static constexpr int FF    = 4096;
static constexpr int HD    = 64;          // head dim
static constexpr int KIN   = 768;         // ceil(0.75*E)  : kept inputs of GEMM1
static constexpr int KOUT  = 3072;        // ceil(0.75*FF) : kept outputs of GEMM1
static constexpr int EOUT  = 768;         // ceil(0.75*E)  : kept outputs of GEMM2
static constexpr int TM    = 32;          // tokens per block
static constexpr int FC    = 128;         // FF chunk width (1 col tile per wave)
static constexpr int NTOK  = 4 * 4096;    // B*S
static constexpr float EPSV = 1e-5f;

union ABFrag {
  uint4 u[2];
  v16bf v;
};

union Pack8 {
  __bf16 h[8];
  uint4  u;
};

__device__ __forceinline__ v8f wmma_bf16(const v16bf a, const v16bf b, const v8f c) {
  // (neg_a, A, neg_b, B, c_mod, C, reuse_a, reuse_b)
  return __builtin_amdgcn_wmma_f32_16x16x32_bf16(false, a, false, b, (short)0, c,
                                                 false, false);
}

// ---------------------------------------------------------------------------
// Pass 1: apply triangular ordered-dropout masks and convert weights to bf16.
// Blocks [0, 2048)  : Wfc  [FF, E]   mask: 4j <= i && j < KIN  && i < KOUT
// Blocks [2048,4096): Wproj[E, FF]   mask: f <= 4e && f < KOUT && e < EOUT
// ---------------------------------------------------------------------------
__global__ __launch_bounds__(256) void prep_weights(
    const float* __restrict__ Wfc, const float* __restrict__ Wproj,
    __bf16* __restrict__ WfcB, __bf16* __restrict__ WprojB)
{
  const long half = (long)FF * E / 8;                 // 512K vec8 per matrix
  long vid = (long)blockIdx.x * 256 + threadIdx.x;    // 0 .. 2*half-1
  const bool isProj = vid >= half;
  if (isProj) vid -= half;
  const long base = vid * 8;

  const float* src = (isProj ? Wproj : Wfc) + base;
  float4 f0 = *(const float4*)(src);
  float4 f1 = *(const float4*)(src + 4);

  int row, col;                                       // row = out unit, col = in unit
  if (isProj) { row = (int)(base / FF); col = (int)(base - (long)row * FF); }
  else        { row = (int)(base / E);  col = (int)(base - (long)row * E);  }

  int bound;                                          // keep col <= bound
  if (isProj) bound = (row < EOUT) ? min(4 * row, KOUT - 1) : -1;
  else        bound = (row < KOUT) ? min(row >> 2, KIN - 1) : -1;

  Pack8 p;
  const float v[8] = {f0.x, f0.y, f0.z, f0.w, f1.x, f1.y, f1.z, f1.w};
#pragma unroll
  for (int q = 0; q < 8; ++q)
    p.h[q] = (__bf16)((col + q <= bound) ? v[q] : 0.0f);

  *(uint4*)((isProj ? WprojB : WfcB) + base) = p.u;
}

// ---------------------------------------------------------------------------
// Pass 2: fused LN + masked MLP. 512 blocks x 256 threads (8 wave32s).
// ---------------------------------------------------------------------------
__global__ __launch_bounds__(256) void triod_mlp_fused(
    const float*  __restrict__ x,       // [NTOK, E]  fp32
    const __bf16* __restrict__ WfcB,    // [FF, E]    masked bf16
    const __bf16* __restrict__ WprojB,  // [E, FF]    masked bf16
    float* __restrict__ y)              // [NTOK, E]  fp32
{
  __shared__ __bf16 xn[TM * KIN];   // LN'd activations, kept inputs only (48 KB)
  __shared__ __bf16 g[TM * FC];     // relu^2 chunk of hidden (8 KB)

  const int tid = threadIdx.x;
  const int w   = tid >> 5;         // wave id, 0..7
  const int l   = tid & 31;         // lane
  const int r16 = l & 15;
  const int hi  = l >> 4;
  const long n0 = (long)blockIdx.x * TM;

  // Wave-uniform ids forced into SGPRs so loop bounds / tile-skip tests are
  // scalar (s_cmp/s_cbranch) instead of exec-mask sequences.
  const int wS    = __builtin_amdgcn_readfirstlane(w);
  const int eBase = wS * 96;        // wave's output column base (scalar)

  // ---- Phase 0: per-head LayerNorm, heads 0..11 only (j >= 768 dead) ----
  constexpr int HKEEP = KIN / HD;   // 12 heads kept
  for (int gi = tid; gi < TM * HKEEP; gi += 256) {
    const int tok  = gi / HKEEP;
    const int head = gi - tok * HKEEP;
    const float* src = x + (n0 + tok) * (long)E + head * HD;
    float s = 0.0f, ss = 0.0f;
#pragma unroll
    for (int d = 0; d < HD; d += 4) {
      float4 f = *(const float4*)(src + d);
      s  += f.x + f.y + f.z + f.w;
      ss += f.x * f.x + f.y * f.y + f.z * f.z + f.w * f.w;
    }
    const float mu = s * (1.0f / HD);
    const float rs = rsqrtf(ss * (1.0f / HD) - mu * mu + EPSV);
    __bf16* dst = &xn[tok * KIN + head * HD];
#pragma unroll
    for (int d = 0; d < HD; d += 4) {
      float4 f = *(const float4*)(src + d);   // L0-hit reload
      dst[d + 0] = (__bf16)((f.x - mu) * rs);
      dst[d + 1] = (__bf16)((f.y - mu) * rs);
      dst[d + 2] = (__bf16)((f.z - mu) * rs);
      dst[d + 3] = (__bf16)((f.w - mu) * rs);
    }
  }
  __syncthreads();

  // ---- Output accumulators: wave w owns y[:, 96w .. 96w+95] (2x6 tiles) ----
  v8f acc[2][6];
#pragma unroll
  for (int a = 0; a < 2; ++a)
#pragma unroll
    for (int b = 0; b < 6; ++b)
#pragma unroll
      for (int q = 0; q < 8; ++q) acc[a][b][q] = 0.0f;

  // A-fragment base pointers (ISA 16-bit A 16x32 lane layout).
  const __bf16* ap0Base = &xn[(0 * 16 + r16) * KIN + hi * 8];
  const __bf16* ap1Base = &xn[(1 * 16 + r16) * KIN + hi * 8];
  const __bf16* gp0     = &g[(0 * 16 + r16) * FC + hi * 8];
  const __bf16* gp1     = &g[(1 * 16 + r16) * FC + hi * 8];

  // ---- Main loop over FF chunks; h[:, f>=KOUT] == 0, skip entirely ----
  for (int f0 = 0; f0 < KOUT; f0 += FC) {
    // GEMM1: wave w owns col tile ct1 = w; computes BOTH token row tiles
    // against the same weight fragments (B loaded once per block).
    const int iBase = f0 + wS * 16;           // scalar tile row base
    const int iCol  = iBase + r16;            // this lane's B column (N)

    // Prefetch next chunk's W_fc rows into GL2 (speculative).
    if (f0 + FC < KOUT)
      __builtin_prefetch(WfcB + (long)(iCol + FC) * E, 0, 1);

    // Triangular K bound (scalar): only j <= (i_max)/4 contribute; round to 64.
    const int jmax  = (iBase + 15) >> 2;
    const int Ktile = min(KIN, (jmax + 1 + 63) & ~63);

    v8f hc0, hc1;
#pragma unroll
    for (int q = 0; q < 8; ++q) { hc0[q] = 0.0f; hc1[q] = 0.0f; }

    const __bf16* ap0 = ap0Base;
    const __bf16* ap1 = ap1Base;
    const __bf16* bp  = WfcB + (long)iCol * E + hi * 8;
    for (int k = 0; k < Ktile; k += 64) {
      ABFrag b0, b1, a00, a01, a10, a11;
      b0.u[0]  = *(const uint4*)(bp);          // K = k   .. k+31
      b0.u[1]  = *(const uint4*)(bp + 16);
      b1.u[0]  = *(const uint4*)(bp + 32);     // K = k+32 .. k+63
      b1.u[1]  = *(const uint4*)(bp + 48);
      a00.u[0] = *(const uint4*)(ap0);
      a00.u[1] = *(const uint4*)(ap0 + 16);
      a01.u[0] = *(const uint4*)(ap0 + 32);
      a01.u[1] = *(const uint4*)(ap0 + 48);
      a10.u[0] = *(const uint4*)(ap1);
      a10.u[1] = *(const uint4*)(ap1 + 16);
      a11.u[0] = *(const uint4*)(ap1 + 32);
      a11.u[1] = *(const uint4*)(ap1 + 48);
      hc0 = wmma_bf16(a00.v, b0.v, hc0);
      hc1 = wmma_bf16(a10.v, b0.v, hc1);
      hc0 = wmma_bf16(a01.v, b1.v, hc0);
      hc1 = wmma_bf16(a11.v, b1.v, hc1);
      ap0 += 64; ap1 += 64; bp += 64;
    }

    // ---- ReLU^2 and stage chunk of g into LDS (bf16) ----
#pragma unroll
    for (int r = 0; r < 8; ++r) {
      float v0 = fmaxf(hc0[r], 0.0f);
      float v1 = fmaxf(hc1[r], 0.0f);
      v0 *= v0; v1 *= v1;
      const int rr  = r + hi * 8;             // C layout: M = r + 8*(lane>=16)
      const int col = wS * 16 + r16;          // N = lane%16
      g[(rr)      * FC + col] = (__bf16)v0;
      g[(rr + 16) * FC + col] = (__bf16)v1;
    }
    __syncthreads();

    // ---- GEMM2: acc += g_chunk @ WprojB[eBase..eBase+95, :]^T ----
    // Load all A fragments once (4 K-steps x 2 row tiles = 16 ds_load_b128).
    ABFrag af[2][4];
#pragma unroll
    for (int kk = 0; kk < 4; ++kk) {
      af[0][kk].u[0] = *(const uint4*)(gp0 + kk * 32);
      af[0][kk].u[1] = *(const uint4*)(gp0 + kk * 32 + 16);
      af[1][kk].u[0] = *(const uint4*)(gp1 + kk * 32);
      af[1][kk].u[1] = *(const uint4*)(gp1 + kk * 32 + 16);
    }
#pragma unroll
    for (int ct2 = 0; ct2 < 6; ++ct2) {
      const int eTileMax = eBase + ct2 * 16 + 15;   // scalar
      if (f0 > 4 * eTileMax) continue;        // scalar skip: tile fully masked
      const int e = eBase + ct2 * 16 + r16;   // lane's output column
      const __bf16* wp = WprojB + (long)e * FF + f0 + hi * 8;
#pragma unroll
      for (int kk = 0; kk < 4; ++kk) {
        ABFrag b;
        b.u[0] = *(const uint4*)(wp + kk * 32);
        b.u[1] = *(const uint4*)(wp + kk * 32 + 16);
        acc[0][ct2] = wmma_bf16(af[0][kk].v, b.v, acc[0][ct2]);
        acc[1][ct2] = wmma_bf16(af[1][kk].v, b.v, acc[1][ct2]);
      }
    }
    __syncthreads();   // protect g before next chunk overwrites it
  }

  // ---- Write-out: columns >= EOUT are exactly zero; rest from acc ----
  for (int idx = tid; idx < TM * (E - EOUT); idx += 256) {
    const int row = idx / (E - EOUT);
    const int col = EOUT + (idx - row * (E - EOUT));
    y[(n0 + row) * (long)E + col] = 0.0f;
  }
#pragma unroll
  for (int mt = 0; mt < 2; ++mt)
#pragma unroll
    for (int ct2 = 0; ct2 < 6; ++ct2)
#pragma unroll
      for (int r = 0; r < 8; ++r) {
        const int row = mt * 16 + r + hi * 8;
        const int col = eBase + ct2 * 16 + r16;
        y[(n0 + row) * (long)E + col] = acc[mt][ct2][r];
      }
}

extern "C" void kernel_launch(void* const* d_in, const int* in_sizes, int n_in,
                              void* d_out, int out_size, void* d_ws, size_t ws_size,
                              hipStream_t stream) {
  (void)in_sizes; (void)n_in; (void)out_size; (void)ws_size;
  const float* x     = (const float*)d_in[0];
  const float* Wfc   = (const float*)d_in[1];
  const float* Wproj = (const float*)d_in[2];
  float* y = (float*)d_out;

  // d_ws layout: [FF*E bf16 WfcB][E*FF bf16 WprojB] = 16 MB
  __bf16* WfcB   = (__bf16*)d_ws;
  __bf16* WprojB = WfcB + (size_t)FF * E;

  // Pass 1: 2 matrices * 4M elements / (256 thr * 8 elem) = 4096 blocks.
  prep_weights<<<dim3(4096), dim3(256), 0, stream>>>(Wfc, Wproj, WfcB, WprojB);

  // Pass 2: fused LN + double GEMM.
  triod_mlp_fused<<<dim3(NTOK / TM), dim3(256), 0, stream>>>(x, WfcB, WprojB, y);
}
